// DGCNN_27547920237122
// MI455X (gfx1250) — matrix operand order repaired and checked by reading, not compiled, not run on hardware
//
#include <hip/hip_runtime.h>

typedef float    v2f  __attribute__((ext_vector_type(2)));
typedef float    v8f  __attribute__((ext_vector_type(8)));
typedef _Float16 v8h  __attribute__((ext_vector_type(8)));
typedef _Float16 v16h __attribute__((ext_vector_type(16)));

static constexpr int BB   = 16;
static constexpr int NN   = 2048;
static constexpr int KNBR = 20;
static constexpr float EPSV = 1e-5f;

// ---------------- workspace offsets (in floats) ----------------
static constexpr size_t OFF_X1   = 0;                       // 16*2048*64
static constexpr size_t OFF_X2   = 2097152;                 // 16*2048*64
static constexpr size_t OFF_X3   = 4194304;                 // 16*2048*128
static constexpr size_t OFF_X4   = 8388608;                 // 16*2048*256
static constexpr size_t OFF_HMAX = 16777216;                // 16*2048*256 (reused as Y)
static constexpr size_t OFF_HMIN = 25165824;                // 16*2048*256
static constexpr size_t OFF_SQ   = 33554432;                // 16*2048
static constexpr size_t OFF_IDX  = 33587200;                // 16*2048*20 ints
static constexpr size_t OFF_SUMS = 34242560;                // 256 sums + 256 sumsq
static constexpr size_t OFF_WH   = 34243072;                // f16 padded weights (<=256KB)

// ---------------- small helpers ----------------
__global__ void zero_kernel(float* p, int n) {
  int i = blockIdx.x * 256 + threadIdx.x;
  if (i < n) p[i] = 0.f;
}

// zero-padded fp32 -> fp16 weight repack: Wh[o][Kp]
__global__ void wconv_kernel(const float* __restrict__ W, _Float16* __restrict__ Wh,
                             int o, int K2, int Kp) {
  int i = blockIdx.x * 256 + threadIdx.x;
  if (i >= o * Kp) return;
  int r = i / Kp, c = i % Kp;
  Wh[i] = (c < K2) ? (_Float16)W[(size_t)r * K2 + c] : (_Float16)0.f;
}

template <int D>
__global__ void sqnorm_kernel(const float* __restrict__ X, float* __restrict__ SQ) {
  int i = blockIdx.x * 256 + threadIdx.x;
  if (i >= BB * NN) return;
  float s = 0.f;
  #pragma unroll
  for (int c = 0; c < D; ++c) { float v = X[(size_t)i * D + c]; s += v * v; }
  SQ[i] = s;
}

// ---------------- distance (fp32 WMMA) + top-20 selection ----------------
// One workgroup: 16 query rows of one batch. Gram tiles via V_WMMA_F32_16X16X4_F32,
// keys = |x_m|^2 - 2 x_n.x_m (row-constant |x_n|^2 dropped; same ordering as ref).
template <int D>
__global__ __launch_bounds__(128)
void dist_topk_kernel(const float* __restrict__ X, const float* __restrict__ SQ,
                      int* __restrict__ IDX) {
  constexpr int DP = (D + 3) & ~3;
  __shared__ float s_arow[16 * DP];
  __shared__ float s_dist[16 * 512];   // 32KB: one 512-column chunk of keys
  __shared__ float s_topv[16 * 20];
  __shared__ int   s_topi[16 * 20];

  const int b    = blockIdx.y;
  const int row0 = blockIdx.x * 16;
  const int tid  = threadIdx.x;
  const int wv   = tid >> 5, lane = tid & 31;
  const int hf   = lane >> 4, lm = lane & 15;

  for (int i = tid; i < 16 * DP; i += 128) {
    int r = i / DP, c = i % DP;
    s_arow[i] = (c < D) ? X[((size_t)(b * NN + row0 + r)) * D + c] : 0.f;
  }
  for (int i = tid; i < 16 * 20; i += 128) { s_topv[i] = 3.0e38f; s_topi[i] = 0; }
  __syncthreads();

  for (int ch = 0; ch < 4; ++ch) {
    const int cb = ch * 512;
    for (int s = wv; s < 32; s += 4) {             // 16-col subtiles of the chunk
      const int col = cb + s * 16 + lm;
      const float* xc = &X[((size_t)(b * NN + col)) * D];
      v8f acc = {};
      #pragma unroll
      for (int k0 = 0; k0 < DP; k0 += 4) {
        const int kk = k0 + hf * 2;                // A: row=lm, K split by half-wave
        v2f a = *(const v2f*)&s_arow[lm * DP + kk];
        v2f bf;
        if constexpr ((D & 3) == 0) {
          bf = *(const v2f*)&xc[kk];               // unconditional b64 load
        } else {
          bf.x = (kk     < D) ? xc[kk]     : 0.f;
          bf.y = (kk + 1 < D) ? xc[kk + 1] : 0.f;
        }
        acc = __builtin_amdgcn_wmma_f32_16x16x4_f32(false, a, false, bf,
                                                    (short)0, acc, false, false);
      }
      const float sqc = SQ[b * NN + col];
      #pragma unroll
      for (int r = 0; r < 8; ++r) {               // C: M = r + 8*hf, N = lm
        s_dist[(r + hf * 8) * 512 + s * 16 + lm] = sqc - 2.0f * acc[r];
      }
    }
    __syncthreads();
    if (tid < 16) {                                // per-row running sorted top-20
      float* tv = &s_topv[tid * 20];
      int*   ti = &s_topi[tid * 20];
      for (int m = 0; m < 512; ++m) {
        float v = s_dist[tid * 512 + m];
        if (v < tv[19]) {
          int p = 19;
          while (p > 0 && tv[p - 1] > v) { tv[p] = tv[p - 1]; ti[p] = ti[p - 1]; --p; }
          tv[p] = v; ti[p] = cb + m;
        }
      }
    }
    __syncthreads();
  }
  for (int i = tid; i < 16 * 20; i += 128) {
    int r = i / 20, k = i % 20;
    IDX[((size_t)(b * NN + row0 + r)) * 20 + k] = s_topi[r * 20 + k];
  }
}

// ---------------- fused edge GEMM (f16 WMMA) + k-max/min + BN stats ----------------
// Workgroup: 8 points (160 rows, complete k-groups) x 64 channels. H kept in LDS.
template <int D, int O>
__global__ __launch_bounds__(128)
void edge_gemm_kernel(const float* __restrict__ X, const int* __restrict__ IDX,
                      const _Float16* __restrict__ Wh, float* __restrict__ HMAX,
                      float* __restrict__ HMIN, float* __restrict__ SUMS,
                      float* __restrict__ SUMSQ) {
  constexpr int Kp = (2 * D + 31) & ~31;
  __shared__ _Float16 s_a[16 * Kp];    // one M-tile of edge features (<=8KB)
  __shared__ float    s_h[160 * 64];   // 40KB: H tile
  __shared__ int      s_j[16];

  const int b   = blockIdx.z;
  const int p0  = blockIdx.x * 8;
  const int ct0 = blockIdx.y * 64;
  const int tid = threadIdx.x;
  const int wv  = tid >> 5, lane = tid & 31;
  const int hf  = lane >> 4, lm = lane & 15;
  const int col = ct0 + wv * 16 + lm;

  for (int mt = 0; mt < 10; ++mt) {
    __syncthreads();                               // protect s_a from prior iter
    if (tid < 16) {
      int gr = mt * 16 + tid;
      s_j[tid] = IDX[((size_t)(b * NN + p0 + gr / 20)) * 20 + (gr % 20)];
    }
    __syncthreads();
    for (int i = tid; i < 16 * Kp; i += 128) {     // edge feat: [x_n, x_j - x_n]
      int r = i / Kp, c = i % Kp;
      int n0 = p0 + (mt * 16 + r) / 20;
      float v;
      if (c < D)           v = X[((size_t)(b * NN + n0)) * D + c];
      else if (c < 2 * D)  v = X[((size_t)(b * NN + s_j[r])) * D + (c - D)]
                             - X[((size_t)(b * NN + n0)) * D + (c - D)];
      else                 v = 0.f;
      s_a[r * Kp + c] = (_Float16)v;
    }
    __syncthreads();
    v8f acc = {};
    #pragma unroll
    for (int kt = 0; kt < Kp; kt += 32) {
      const int ab = hf * 8;                       // A f16 layout: K halves by half-wave
      v8h alo = *(const v8h*)&s_a[lm * Kp + kt + ab];
      v8h ahi = *(const v8h*)&s_a[lm * Kp + kt + 16 + ab];
      v16h af = __builtin_shufflevector(alo, ahi, 0, 1, 2, 3, 4, 5, 6, 7,
                                        8, 9, 10, 11, 12, 13, 14, 15);
      const int bb0 = hf * 8;                      // B: col=lm, K striped over VGPRs
      v8h blo = *(const v8h*)&Wh[(size_t)col * Kp + kt + bb0];
      v8h bhi = *(const v8h*)&Wh[(size_t)col * Kp + kt + 16 + bb0];
      v16h bf = __builtin_shufflevector(blo, bhi, 0, 1, 2, 3, 4, 5, 6, 7,
                                        8, 9, 10, 11, 12, 13, 14, 15);
      acc = __builtin_amdgcn_wmma_f32_16x16x32_f16(false, af, false, bf,
                                                   (short)0, acc, false, false);
    }
    #pragma unroll
    for (int r = 0; r < 8; ++r)
      s_h[(mt * 16 + r + hf * 8) * 64 + wv * 16 + lm] = acc[r];
  }
  __syncthreads();
  if (tid < 64) {                                  // per-channel: max/min over k + stats
    const int c = tid;
    float s = 0.f, s2 = 0.f;
    for (int p = 0; p < 8; ++p) {
      float mx = -3.0e38f, mn = 3.0e38f;
      #pragma unroll
      for (int k = 0; k < 20; ++k) {
        float v = s_h[(p * 20 + k) * 64 + c];
        s += v; s2 += v * v;
        mx = fmaxf(mx, v); mn = fminf(mn, v);
      }
      HMAX[((size_t)(b * NN + p0 + p)) * O + ct0 + c] = mx;
      HMIN[((size_t)(b * NN + p0 + p)) * O + ct0 + c] = mn;
    }
    atomicAdd(&SUMS[ct0 + c], s);
    atomicAdd(&SUMSQ[ct0 + c], s2);
  }
}

// BN+ReLU applied to the k-extremum (monotone in h; pick max/min by sign of scale).
__global__ void edge_bn_kernel(const float* __restrict__ HMAX, const float* __restrict__ HMIN,
                               const float* __restrict__ G, const float* __restrict__ Bt,
                               const float* __restrict__ SUMS, const float* __restrict__ SUMSQ,
                               float* __restrict__ XOUT, int o) {
  size_t e = (size_t)blockIdx.x * 256 + threadIdx.x;
  if (e >= (size_t)BB * NN * o) return;
  int c = (int)(e % o);
  const float cnt = (float)BB * NN * KNBR;
  float mean = SUMS[c] / cnt;
  float var  = SUMSQ[c] / cnt - mean * mean;
  float sc   = G[c] * rsqrtf(var + EPSV);
  float h    = (sc >= 0.f) ? HMAX[e] : HMIN[e];
  XOUT[e] = fmaxf(sc * (h - mean) + Bt[c], 0.f);
}

// ---------------- final 512->256 GEMM over concat(x1..x4), with BN stats ----------------
__global__ __launch_bounds__(128)
void final_gemm_kernel(const float* __restrict__ X1, const float* __restrict__ X2,
                       const float* __restrict__ X3, const float* __restrict__ X4,
                       const _Float16* __restrict__ W5h, float* __restrict__ Y,
                       float* __restrict__ SUMS, float* __restrict__ SUMSQ) {
  __shared__ _Float16 s_a[16 * 512];
  const int r0  = blockIdx.x * 16;                 // flat (b*N+n) row base
  const int ct0 = blockIdx.y * 64;
  const int tid = threadIdx.x;
  const int wv  = tid >> 5, lane = tid & 31, hf = lane >> 4, lm = lane & 15;
  const int col = ct0 + wv * 16 + lm;

  for (int i = tid; i < 16 * 512; i += 128) {
    int r = i >> 9, c = i & 511;
    size_t gp = (size_t)(r0 + r);
    float v;
    if (c < 64)       v = X1[gp * 64 + c];
    else if (c < 128) v = X2[gp * 64 + (c - 64)];
    else if (c < 256) v = X3[gp * 128 + (c - 128)];
    else              v = X4[gp * 256 + (c - 256)];
    s_a[i] = (_Float16)v;
  }
  __syncthreads();
  v8f acc = {};
  #pragma unroll
  for (int kt = 0; kt < 512; kt += 32) {
    const int ab = hf * 8;
    v8h alo = *(const v8h*)&s_a[lm * 512 + kt + ab];
    v8h ahi = *(const v8h*)&s_a[lm * 512 + kt + 16 + ab];
    v16h af = __builtin_shufflevector(alo, ahi, 0, 1, 2, 3, 4, 5, 6, 7,
                                      8, 9, 10, 11, 12, 13, 14, 15);
    const int bb0 = hf * 8;
    v8h blo = *(const v8h*)&W5h[(size_t)col * 512 + kt + bb0];
    v8h bhi = *(const v8h*)&W5h[(size_t)col * 512 + kt + 16 + bb0];
    v16h bf = __builtin_shufflevector(blo, bhi, 0, 1, 2, 3, 4, 5, 6, 7,
                                      8, 9, 10, 11, 12, 13, 14, 15);
    acc = __builtin_amdgcn_wmma_f32_16x16x32_f16(false, af, false, bf,
                                                 (short)0, acc, false, false);
  }
  float s = 0.f, s2 = 0.f;
  #pragma unroll
  for (int r = 0; r < 8; ++r) {
    float v = acc[r];
    Y[(size_t)(r0 + r + hf * 8) * 256 + col] = v;
    s += v; s2 += v * v;
  }
  atomicAdd(&SUMS[col], s);
  atomicAdd(&SUMSQ[col], s2);
}

__global__ void final_bn_kernel(const float* __restrict__ Y, const float* __restrict__ G,
                                const float* __restrict__ Bt, const float* __restrict__ SUMS,
                                const float* __restrict__ SUMSQ, float* __restrict__ OUT) {
  size_t e = (size_t)blockIdx.x * 256 + threadIdx.x;
  if (e >= (size_t)BB * NN * 256) return;
  int c = (int)(e % 256);
  size_t bn = e / 256;
  int b = (int)(bn / NN), n = (int)(bn % NN);
  const float cnt = (float)BB * NN;
  float mean = SUMS[c] / cnt;
  float var  = SUMSQ[c] / cnt - mean * mean;
  float v = G[c] * (Y[e] - mean) * rsqrtf(var + EPSV) + Bt[c];
  OUT[((size_t)b * 256 + c) * NN + n] = fmaxf(v, 0.f);   // (B, C, N)
}

// ---------------- per-layer driver (templated) ----------------
template <int D, int O>
static void run_layer(const float* xin, const float* W, const float* g, const float* bt,
                      float* xout, float* sq, int* idx, float* hmax, float* hmin,
                      float* sums, float* sumsq, _Float16* wh, hipStream_t stream) {
  constexpr int Kp = (2 * D + 31) & ~31;
  sqnorm_kernel<D><<<(BB * NN) / 256, 256, 0, stream>>>(xin, sq);
  dist_topk_kernel<D><<<dim3(NN / 16, BB), 128, 0, stream>>>(xin, sq, idx);
  wconv_kernel<<<(O * Kp + 255) / 256, 256, 0, stream>>>(W, wh, O, 2 * D, Kp);
  zero_kernel<<<2, 256, 0, stream>>>(sums, 512);
  edge_gemm_kernel<D, O><<<dim3(NN / 8, O / 64, BB), 128, 0, stream>>>(
      xin, idx, wh, hmax, hmin, sums, sumsq);
  edge_bn_kernel<<<(BB * NN * O + 255) / 256, 256, 0, stream>>>(
      hmax, hmin, g, bt, sums, sumsq, xout, O);
}

// ---------------- driver ----------------
extern "C" void kernel_launch(void* const* d_in, const int* in_sizes, int n_in,
                              void* d_out, int out_size, void* d_ws, size_t ws_size,
                              hipStream_t stream) {
  const float* x  = (const float*)d_in[0];
  const float* W1 = (const float*)d_in[1];  const float* g1 = (const float*)d_in[2];  const float* b1 = (const float*)d_in[3];
  const float* W2 = (const float*)d_in[4];  const float* g2 = (const float*)d_in[5];  const float* b2 = (const float*)d_in[6];
  const float* W3 = (const float*)d_in[7];  const float* g3 = (const float*)d_in[8];  const float* b3 = (const float*)d_in[9];
  const float* W4 = (const float*)d_in[10]; const float* g4 = (const float*)d_in[11]; const float* b4 = (const float*)d_in[12];
  const float* W5 = (const float*)d_in[13]; const float* g5 = (const float*)d_in[14]; const float* b5 = (const float*)d_in[15];

  float* ws    = (float*)d_ws;
  float* x1    = ws + OFF_X1;
  float* x2    = ws + OFF_X2;
  float* x3    = ws + OFF_X3;
  float* x4    = ws + OFF_X4;
  float* hmax  = ws + OFF_HMAX;
  float* hmin  = ws + OFF_HMIN;
  float* y     = ws + OFF_HMAX;                    // reuse after layer 4
  float* sq    = ws + OFF_SQ;
  int*   idx   = (int*)(ws + OFF_IDX);
  float* sums  = ws + OFF_SUMS;                    // [256] sums, [256] sumsq
  float* sumsq = sums + 256;
  _Float16* wh = (_Float16*)(ws + OFF_WH);

  run_layer<3, 64>(x,  W1, g1, b1, x1, sq, idx, hmax, hmin, sums, sumsq, wh, stream);
  run_layer<64, 64>(x1, W2, g2, b2, x2, sq, idx, hmax, hmin, sums, sumsq, wh, stream);
  run_layer<64, 128>(x2, W3, g3, b3, x3, sq, idx, hmax, hmin, sums, sumsq, wh, stream);
  run_layer<128, 256>(x3, W4, g4, b4, x4, sq, idx, hmax, hmin, sums, sumsq, wh, stream);

  wconv_kernel<<<(256 * 512 + 255) / 256, 256, 0, stream>>>(W5, wh, 256, 512, 512);
  zero_kernel<<<2, 256, 0, stream>>>(sums, 512);
  final_gemm_kernel<<<dim3((BB * NN) / 16, 4), 128, 0, stream>>>(
      x1, x2, x3, x4, wh, y, sums, sumsq);
  final_bn_kernel<<<(BB * NN * 256) / 256, 256, 0, stream>>>(
      y, g5, b5, sums, sumsq, (float*)d_out);
}
